// RegionDecouplingLoss_10780367912988
// MI455X (gfx1250) — matrix operand order repaired and checked by reading, not compiled, not run on hardware
//
#include <hip/hip_runtime.h>
#include <hip/hip_bf16.h>

typedef __attribute__((ext_vector_type(16))) _Float16 v16h;
typedef __attribute__((ext_vector_type(8)))  float    v8f;

#define B_   16
#define H_   256
#define W_   256
#define N_   65536
#define BN_  1048576
#define LSTR  272          // padded LDS row stride (floats), multiple of 16
#define LROWS 258          // 256 rows + zero top/bottom
#define LDS_FLOATS (LSTR * LROWS)   // 70176 floats = 280704 B <= 320 KB
#define SDT_ITERS (H_ + W_)         // 512
#define BIGV 1e6f

// ---------------- small helpers ----------------

__device__ __forceinline__ void gauss9(float* gn) {
  float g[9]; float s = 0.f;
#pragma unroll
  for (int i = 0; i < 9; ++i) { float d = (float)(i - 4); g[i] = __expf(-(d * d) / 4.5f); s += g[i]; }
#pragma unroll
  for (int i = 0; i < 9; ++i) gn[i] = g[i] / s;
}

__device__ __forceinline__ float sigmoidf_(float x) { return 1.0f / (1.0f + __expf(-x)); }

// ---------------- dilation (5x5 max, separable, SAME/-inf pad) ----------------

__global__ __launch_bounds__(256) void dilate_h_k(const float* __restrict__ Y, float* __restrict__ T) {
  int i = blockIdx.x * 256 + threadIdx.x;          // [0, BN)
  int c = i & 255; int rb = i >> 8;                // rb = b*256 + r
  float mx = -1e30f;
#pragma unroll
  for (int d = -2; d <= 2; ++d) {
    int cc = c + d;
    if (cc >= 0 && cc < W_) {
      float v = Y[rb * W_ + cc];
      v = fminf(fmaxf(v, 0.0f), 1.0f);             // clip(Y,0,1)
      mx = fmaxf(mx, v);
    }
  }
  T[i] = mx;
}

__global__ __launch_bounds__(256) void dilate_v_k(const float* __restrict__ T, _Float16* __restrict__ Yd) {
  int i = blockIdx.x * 256 + threadIdx.x;
  int c = i & 255; int r = (i >> 8) & 255; int b = i >> 16;
  float mx = -1e30f;
#pragma unroll
  for (int d = -2; d <= 2; ++d) {
    int rr = r + d;
    if (rr >= 0 && rr < H_) mx = fmaxf(mx, T[(b * H_ + rr) * W_ + c]);
  }
  Yd[i] = (_Float16)mx;
}

// ---------------- Gaussian blur via WMMA (separable 9-tap, zero pad) ----------------
// Row pass: D(16x16) = A(16x32 image rows) x B(32x16 banded gaussian), one wmma/tile.

__global__ __launch_bounds__(256) void blur_row_wmma_k(const _Float16* __restrict__ X, _Float16* __restrict__ O) {
  float gn[9]; gauss9(gn);
  const int wave = threadIdx.x >> 5;
  const int lane = threadIdx.x & 31;
  const int tileId = blockIdx.x * 8 + wave;        // 4096 tiles total
  const int img = tileId >> 8;
  const int t = tileId & 255;
  const int r0 = (t >> 4) * 16;
  const int c0 = (t & 15) * 16;
  const int m = lane & 15, half = lane >> 4;
  const _Float16* base = X + (size_t)img * N_;

  v16h a;                                          // A[m][K] = X[r0+m][c0-8+K]
#pragma unroll
  for (int e = 0; e < 16; ++e) {
    int K = (e < 8) ? (half * 8 + e) : (16 + half * 8 + (e - 8));
    int col = c0 - 8 + K;
    a[e] = (col >= 0 && col < W_) ? base[(r0 + m) * W_ + col] : (_Float16)0.0f;
  }
  v16h bm;                                         // B[K][j] = g[K-j-4], j = lane&15
#pragma unroll
  for (int e = 0; e < 16; ++e) {
    int K = (e < 8) ? (half * 8 + e) : (16 + half * 8 + (e - 8));
    int tw = K - (lane & 15) - 4;
    bm[e] = (tw >= 0 && tw <= 8) ? (_Float16)gn[tw] : (_Float16)0.0f;
  }
  v8f c = {};
  c = __builtin_amdgcn_wmma_f32_16x16x32_f16(false, a, false, bm, (short)0, c, false, false);
  _Float16* ob = O + (size_t)img * N_;
#pragma unroll
  for (int v = 0; v < 8; ++v) {
    int row = r0 + v + 8 * half;
    int col = c0 + (lane & 15);
    ob[row * W_ + col] = (_Float16)c[v];
  }
}

// Column pass: D(16x16) = A(16x32 banded gaussian) x B(32x16 image cols), f32 out.
__global__ __launch_bounds__(256) void blur_col_wmma_k(const _Float16* __restrict__ X, float* __restrict__ O) {
  float gn[9]; gauss9(gn);
  const int wave = threadIdx.x >> 5;
  const int lane = threadIdx.x & 31;
  const int tileId = blockIdx.x * 8 + wave;
  const int img = tileId >> 8;
  const int t = tileId & 255;
  const int r0 = (t >> 4) * 16;
  const int c0 = (t & 15) * 16;
  const int m = lane & 15, half = lane >> 4;
  const _Float16* base = X + (size_t)img * N_;

  v16h a;                                          // A[m][K] = g[K-m-4]
#pragma unroll
  for (int e = 0; e < 16; ++e) {
    int K = (e < 8) ? (half * 8 + e) : (16 + half * 8 + (e - 8));
    int tw = K - m - 4;
    a[e] = (tw >= 0 && tw <= 8) ? (_Float16)gn[tw] : (_Float16)0.0f;
  }
  v16h bm;                                         // B[K][j] = X[r0-8+K][c0+j]
#pragma unroll
  for (int e = 0; e < 16; ++e) {
    int K = (e < 8) ? (half * 8 + e) : (16 + half * 8 + (e - 8));
    int row = r0 - 8 + K;
    bm[e] = (row >= 0 && row < H_) ? base[row * W_ + c0 + (lane & 15)] : (_Float16)0.0f;
  }
  v8f c = {};
  c = __builtin_amdgcn_wmma_f32_16x16x32_f16(false, a, false, bm, (short)0, c, false, false);
  float* ob = O + (size_t)img * N_;
#pragma unroll
  for (int v = 0; v < 8; ++v) {
    int row = r0 + v + 8 * half;
    ob[row * W_ + c0 + (lane & 15)] = c[v];
  }
}

// ---------------- per-image min/max + normalize -> soft target ----------------

__global__ __launch_bounds__(256) void minmax_k(const float* __restrict__ ys, float* __restrict__ mnmx) {
  __shared__ float smn[256], smx[256];
  int b = blockIdx.x, t = threadIdx.x;
  float mn = 1e30f, mx = -1e30f;
  for (int i = t; i < N_; i += 256) {
    float v = ys[(size_t)b * N_ + i];
    mn = fminf(mn, v); mx = fmaxf(mx, v);
  }
  smn[t] = mn; smx[t] = mx; __syncthreads();
  for (int s = 128; s > 0; s >>= 1) {
    if (t < s) { smn[t] = fminf(smn[t], smn[t + s]); smx[t] = fmaxf(smx[t], smx[t + s]); }
    __syncthreads();
  }
  if (t == 0) { mnmx[b] = smn[0]; mnmx[16 + b] = smx[0]; }
}

__global__ __launch_bounds__(256) void normalize_k(float* __restrict__ ys, const float* __restrict__ mnmx) {
  int i = blockIdx.x * 256 + threadIdx.x;
  int b = i >> 16;
  float mn = mnmx[b], mx = mnmx[16 + b];
  ys[i] = (ys[i] - mn) / (mx - mn + 1e-8f);
}

// ---------------- persistent SDT (chamfer) : one (image,side) per workgroup ----------------
// LDS layout: padded 258 x 272, value at (row r, col c) -> L[(r+1)*LSTR + c + 4]; border stays 0.

__device__ __forceinline__ void row_h3(const float* __restrict__ Ls, int pr, int c0,
                                       float* __restrict__ h, float* __restrict__ cen) {
  int base = pr * LSTR + c0 + 4;                   // 16B aligned (pr*272 + c0 mult of 16)
  float x[24];
#pragma unroll
  for (int g = 0; g < 6; ++g) {
    float4 v = *reinterpret_cast<const float4*>(&Ls[base - 4 + 4 * g]);
    x[4 * g + 0] = v.x; x[4 * g + 1] = v.y; x[4 * g + 2] = v.z; x[4 * g + 3] = v.w;
  }
#pragma unroll
  for (int j = 0; j < 16; ++j) { h[j] = x[j + 3] + x[j + 4] + x[j + 5]; cen[j] = x[j + 4]; }
}

__global__ __launch_bounds__(1024) void sdt_k(const float* __restrict__ Y, float* __restrict__ D) {
  extern __shared__ float L[];
  const int wg = blockIdx.x;                       // 0..31
  const int b = wg & 15;
  const int side = wg >> 4;                        // 0: din, 1: dout
  const int tid = threadIdx.x;

  for (int i = tid; i < LDS_FLOATS; i += 1024) L[i] = 0.0f;  // zero incl. border ring
  __syncthreads();

  const int r0 = (tid >> 4) * 4;                   // 4 rows x 16 cols per thread
  const int c0 = (tid & 15) * 16;
  const float* src = Y + (size_t)b * N_;

  // async global -> LDS copy of raw mask into padded interior
#pragma unroll
  for (int rr = 0; rr < 4; ++rr) {
    int r = r0 + rr;
#pragma unroll
    for (int cc = 0; cc < 16; cc += 4) {
      unsigned loff = (unsigned)(size_t)(&L[(r + 1) * LSTR + c0 + cc + 4]);
      unsigned long long gaddr = (unsigned long long)(size_t)(src + r * W_ + c0 + cc);
      asm volatile("global_load_async_to_lds_b128 %0, %1, off" :: "v"(loff), "v"(gaddr) : "memory");
    }
  }
  asm volatile("s_wait_asynccnt 0" ::: "memory");
  // transform own pixels in place: din = mask>0.5 ? 0 : BIG ; dout = mask<=0.5 ? 0 : BIG
#pragma unroll
  for (int rr = 0; rr < 4; ++rr) {
#pragma unroll
    for (int cc = 0; cc < 16; ++cc) {
      int si = (r0 + rr + 1) * LSTR + c0 + cc + 4;
      float y = L[si];
      float v = (side == 0) ? ((y > 0.5f) ? 0.0f : BIGV)
                            : ((y <= 0.5f) ? 0.0f : BIGV);
      L[si] = v;
    }
  }
  __syncthreads();

  float nd[64];
  for (int it = 0; it < SDT_ITERS; ++it) {
    float Ha[16], Hb[16], Hc[16], cen[16], cnew[16];
    row_h3(L, r0 + 0, c0, Ha, cnew);               // row r0-1 (cnew unused)
    row_h3(L, r0 + 1, c0, Hb, cen);                // row r0
#pragma unroll
    for (int rr = 0; rr < 4; ++rr) {
      row_h3(L, r0 + 2 + rr, c0, Hc, cnew);        // row r0+1+rr
#pragma unroll
      for (int j = 0; j < 16; ++j) {
        float s = Ha[j] + Hb[j] + Hc[j] - cen[j];  // 8-neighbor sum (zero-padded)
        float cand = s * 0.125f + 1.0f;
        nd[rr * 16 + j] = fminf(cen[j], cand);
      }
#pragma unroll
      for (int j = 0; j < 16; ++j) { Ha[j] = Hb[j]; Hb[j] = Hc[j]; cen[j] = cnew[j]; }
    }
    __syncthreads();                               // all reads done
#pragma unroll
    for (int rr = 0; rr < 4; ++rr) {
      int si = (r0 + rr + 1) * LSTR + c0 + 4;
#pragma unroll
      for (int j = 0; j < 16; j += 4) {
        float4 v = make_float4(nd[rr * 16 + j], nd[rr * 16 + j + 1],
                               nd[rr * 16 + j + 2], nd[rr * 16 + j + 3]);
        *reinterpret_cast<float4*>(&L[si + j]) = v;
      }
    }
    __syncthreads();                               // all writes visible
  }

  // async LDS -> global writeback of the final distance field
  float* dst = D + (size_t)wg * N_;
#pragma unroll
  for (int rr = 0; rr < 4; ++rr) {
    int r = r0 + rr;
#pragma unroll
    for (int cc = 0; cc < 16; cc += 4) {
      unsigned loff = (unsigned)(size_t)(&L[(r + 1) * LSTR + c0 + cc + 4]);
      unsigned long long gaddr = (unsigned long long)(size_t)(dst + r * W_ + c0 + cc);
      asm volatile("global_store_async_from_lds_b128 %0, %1, off" :: "v"(gaddr), "v"(loff) : "memory");
    }
  }
  asm volatile("s_wait_asynccnt 0" ::: "memory");
}

// ---------------- boundary weight ----------------

__global__ __launch_bounds__(256) void bw_k(const float* __restrict__ D, float* __restrict__ BW) {
  int i = blockIdx.x * 256 + threadIdx.x;
  float sdt = D[BN_ + i] - D[i];                   // dout - din
  BW[i] = __expf(-(sdt * sdt) / 200.0f) + 1e-3f;   // 2*gamma^2 = 200
}

// ---------------- stage losses (two-pass deterministic reduction) ----------------

__global__ __launch_bounds__(256) void loss0_k(const float* __restrict__ M0, const float* __restrict__ soft,
                                               const float* __restrict__ BW, float* __restrict__ part) {
  __shared__ float sb[256];
  int t = threadIdx.x;
  int base = blockIdx.x * 1024;
  float acc = 0.0f;
#pragma unroll
  for (int k = 0; k < 4; ++k) {
    int i = base + k * 256 + t;
    float mi = sigmoidf_(M0[i]);
    acc += BW[i] * fabsf(mi - soft[i]);
  }
  sb[t] = acc; __syncthreads();
  for (int s = 128; s > 0; s >>= 1) { if (t < s) sb[t] += sb[t + s]; __syncthreads(); }
  if (t == 0) part[blockIdx.x] = sb[0];
}

__global__ __launch_bounds__(256) void loss1_k(const float* __restrict__ M1, const float* __restrict__ soft,
                                               const float* __restrict__ BW, float* __restrict__ part) {
  __shared__ float sb[256];
  int t = threadIdx.x;
  int base = blockIdx.x * 1024;
  float acc = 0.0f;
#pragma unroll
  for (int k = 0; k < 4; ++k) {
    int i = base + k * 256 + t;                    // [0, 16*128*128)
    int b = i >> 14;
    int rem = i & 16383;
    int oy = rem >> 7, ox = rem & 127;
    size_t p = ((size_t)b << 16) + (size_t)(2 * oy) * W_ + 2 * ox;
    // bilinear 256->128, align_corners=False == exact 2x2 mean (vertical then horizontal)
    float s00 = soft[p], s01 = soft[p + 1], s10 = soft[p + W_], s11 = soft[p + W_ + 1];
    float sp = (s00 * 0.5f + s10 * 0.5f) * 0.5f + (s01 * 0.5f + s11 * 0.5f) * 0.5f;
    float b00 = BW[p], b01 = BW[p + 1], b10 = BW[p + W_], b11 = BW[p + W_ + 1];
    float bp = (b00 * 0.5f + b10 * 0.5f) * 0.5f + (b01 * 0.5f + b11 * 0.5f) * 0.5f;
    float mi = sigmoidf_(M1[i]);
    acc += bp * fabsf(mi - sp);
  }
  sb[t] = acc; __syncthreads();
  for (int s = 128; s > 0; s >>= 1) { if (t < s) sb[t] += sb[t + s]; __syncthreads(); }
  if (t == 0) part[blockIdx.x] = sb[0];
}

__global__ __launch_bounds__(256) void final_k(const float* __restrict__ p0, const float* __restrict__ p1,
                                               float* __restrict__ out) {
  __shared__ float s0[256], s1[256];
  int t = threadIdx.x;
  s0[t] = p0[t] + p0[256 + t] + p0[512 + t] + p0[768 + t];
  s1[t] = p1[t];
  __syncthreads();
  for (int s = 128; s > 0; s >>= 1) {
    if (t < s) { s0[t] += s0[t + s]; s1[t] += s1[t + s]; }
    __syncthreads();
  }
  if (t == 0) {
    float l0 = s0[0] / (16.0f * 65536.0f);
    float l1 = s1[0] / (16.0f * 16384.0f);
    out[0] = 0.5f * (l0 + l1);
  }
}

// ---------------- launcher ----------------

extern "C" void kernel_launch(void* const* d_in, const int* in_sizes, int n_in,
                              void* d_out, int out_size, void* d_ws, size_t ws_size,
                              hipStream_t stream) {
  const float* M0 = (const float*)d_in[0];   // [16,1,256,256]
  const float* M1 = (const float*)d_in[1];   // [16,1,128,128]
  const float* Y  = (const float*)d_in[2];   // [16,1,256,256]
  float* out = (float*)d_out;

  float* ws = (float*)d_ws;
  float*     tmp  = ws;                                  // BN  (later reused as bw)
  _Float16*  yd   = (_Float16*)(ws + BN_);               // BN halfs
  _Float16*  bh   = (_Float16*)(ws + BN_ + BN_ / 2);     // BN halfs
  float*     ys   = ws + 2 * BN_;                        // BN  (soft, in-place normalized)
  float*     mnmx = ws + 3 * BN_;                        // 32
  float*     dsdt = ws + 3 * BN_ + 64;                   // 2*BN
  float*     part0 = ws + 5 * BN_ + 64;                  // 1024
  float*     part1 = part0 + 1024;                       // 256
  float*     bw   = tmp;

  dilate_h_k<<<BN_ / 256, 256, 0, stream>>>(Y, tmp);
  dilate_v_k<<<BN_ / 256, 256, 0, stream>>>(tmp, yd);
  blur_row_wmma_k<<<512, 256, 0, stream>>>(yd, bh);      // 4096 tiles, 1 wmma/wave
  blur_col_wmma_k<<<512, 256, 0, stream>>>(bh, ys);
  minmax_k<<<16, 256, 0, stream>>>(ys, mnmx);
  normalize_k<<<BN_ / 256, 256, 0, stream>>>(ys, mnmx);  // ys -> soft
  sdt_k<<<32, 1024, LDS_FLOATS * sizeof(float), stream>>>(Y, dsdt);
  bw_k<<<BN_ / 256, 256, 0, stream>>>(dsdt, bw);
  loss0_k<<<1024, 256, 0, stream>>>(M0, ys, bw, part0);
  loss1_k<<<256, 256, 0, stream>>>(M1, ys, bw, part1);
  final_k<<<1, 256, 0, stream>>>(part0, part1, out);
}
